// TransformerDecoderLayer_42846593744915
// MI455X (gfx1250) — matrix operand order repaired and checked by reading, not compiled, not run on hardware
//
#include <hip/hip_runtime.h>

// ---------- types ----------
typedef __attribute__((ext_vector_type(16))) __bf16 v16bf;
typedef __attribute__((ext_vector_type(8)))  float  v8f;
typedef __attribute__((ext_vector_type(8)))  unsigned short us8;
typedef __attribute__((ext_vector_type(4)))  unsigned short us4;
typedef __attribute__((ext_vector_type(4)))  int i4;

union FragBF { unsigned int u[8]; v16bf v; };

__device__ __forceinline__ unsigned short f2bf(float f) {
  unsigned int u = __float_as_uint(f);
  u += 0x7FFFu + ((u >> 16) & 1u);     // round-to-nearest-even
  return (unsigned short)(u >> 16);
}
__device__ __forceinline__ float bf2f(unsigned short u) {
  return __uint_as_float((unsigned int)u << 16);
}

// ---------- CDNA5 async global->LDS staging (builtin confirmed present) ----
#if defined(__gfx1250__) && __has_builtin(__builtin_amdgcn_global_load_async_to_lds_b128) && __has_builtin(__builtin_amdgcn_s_wait_asynccnt)
#define USE_ASYNC 1
#else
#define USE_ASYNC 0
#endif

#if USE_ASYNC
typedef __attribute__((address_space(1))) i4 as1_i4;
typedef __attribute__((address_space(3))) i4 as3_i4;
#define ASYNC_CP16(gp, lp) \
  __builtin_amdgcn_global_load_async_to_lds_b128((as1_i4*)(gp), (as3_i4*)(lp), 0, 0)
#define ASYNC_WAIT() __builtin_amdgcn_s_wait_asynccnt(0)
#else
#define ASYNC_CP16(gp, lp) (*(us8*)(lp) = *(const us8*)(gp))
#define ASYNC_WAIT() ((void)0)
#endif

// =====================================================================
// Weight transpose+convert: W fp32 [K][N] -> Wt bf16 [N][K]
// grid (N/64, K/64), block 256. One-time pass; amortized over 64 M-tiles.
// =====================================================================
__global__ __launch_bounds__(256)
void transpose_bf16_kernel(const float* __restrict__ W, unsigned short* __restrict__ Wt,
                           int K, int N) {
  __shared__ unsigned short Ls[64 * 72];
  const int n0 = blockIdx.x * 64;
  const int k0 = blockIdx.y * 64;
  const int t = threadIdx.x;
  {
    const int c  = (t & 15) * 4;
    const int r0 = t >> 4;                        // 0..15
    for (int p = 0; p < 4; ++p) {
      const int k = r0 + p * 16;
      const float4 v4 = *(const float4*)(W + (size_t)(k0 + k) * N + n0 + c);
      Ls[(c + 0) * 72 + k] = f2bf(v4.x);
      Ls[(c + 1) * 72 + k] = f2bf(v4.y);
      Ls[(c + 2) * 72 + k] = f2bf(v4.z);
      Ls[(c + 3) * 72 + k] = f2bf(v4.w);
    }
  }
  __syncthreads();
  {
    const int n = t >> 2;
    const int q = t & 3;
    unsigned short* dst = Wt + (size_t)(n0 + n) * K + k0 + q * 16;
    *(us8*)dst       = *(const us8*)&Ls[n * 72 + q * 16];
    *(us8*)(dst + 8) = *(const us8*)&Ls[n * 72 + q * 16 + 8];
  }
}

// =====================================================================
// LayerNorm: fp32 in -> bf16 out, one block per row of 1024
// =====================================================================
__global__ __launch_bounds__(256)
void ln_kernel(const float* __restrict__ in, const float* __restrict__ g,
               const float* __restrict__ b, unsigned short* __restrict__ out) {
  __shared__ float s1[256];
  __shared__ float s2[256];
  const int row = blockIdx.x;
  const int t = threadIdx.x;
  const float4 v = *(const float4*)(in + (size_t)row * 1024 + t * 4);
  float s  = v.x + v.y + v.z + v.w;
  float ss = v.x * v.x + v.y * v.y + v.z * v.z + v.w * v.w;
  s1[t] = s; s2[t] = ss;
  __syncthreads();
  for (int off = 128; off > 0; off >>= 1) {
    if (t < off) { s1[t] += s1[t + off]; s2[t] += s2[t + off]; }
    __syncthreads();
  }
  const float mu  = s1[0] * (1.0f / 1024.0f);
  const float var = s2[0] * (1.0f / 1024.0f) - mu * mu;
  const float inv = rsqrtf(var + 1e-5f);
  const float4 gg = *(const float4*)(g + t * 4);
  const float4 bb = *(const float4*)(b + t * 4);
  us4 o;
  o[0] = f2bf((v.x - mu) * inv * gg.x + bb.x);
  o[1] = f2bf((v.y - mu) * inv * gg.y + bb.y);
  o[2] = f2bf((v.z - mu) * inv * gg.z + bb.z);
  o[3] = f2bf((v.w - mu) * inv * gg.w + bb.w);
  *(us4*)(out + (size_t)row * 1024 + t * 4) = o;
}

// =====================================================================
// WMMA bf16 GEMM: C[M,N] = A[M,K] * Bt[N,K]^T  (+ R optional, fp32)
// A, Bt bf16 in global. 128x128x32 tiles, double-buffered LDS, async
// global->LDS staging (ASYNCcnt). 8 waves; wave = 2x4 WMMA tiles.
// =====================================================================
#define GLDA 40
#define GLDB 40

__device__ __forceinline__
void gemm_stage(const unsigned short* __restrict__ A, const unsigned short* __restrict__ Bt,
                unsigned short* As, unsigned short* Bs,
                int tid, int blockM, int blockN, int K, int k0) {
  const int row  = tid >> 1;         // 0..127
  const int half = tid & 1;          // 16-element half of the 32-wide K slice
  const unsigned short* ga = A  + (size_t)(blockM + row) * K + k0 + half * 16;
  const unsigned short* gb = Bt + (size_t)(blockN + row) * K + k0 + half * 16;
  unsigned short* la = &As[row * GLDA + half * 16];
  unsigned short* lb = &Bs[row * GLDB + half * 16];
  ASYNC_CP16(ga,     la);
  ASYNC_CP16(ga + 8, la + 8);
  ASYNC_CP16(gb,     lb);
  ASYNC_CP16(gb + 8, lb + 8);
}

__global__ __launch_bounds__(256)
void gemm_kernel(const unsigned short* __restrict__ A, const unsigned short* __restrict__ Bt,
                 void* __restrict__ Cv, const float* __restrict__ R,
                 int N, int Kdim, int cbf16) {
  __shared__ unsigned short As[2][128 * GLDA];
  __shared__ unsigned short Bs[2][128 * GLDB];
  const int tid  = threadIdx.x;
  const int lane = tid & 31;
  const int wave = tid >> 5;
  const int wm = wave & 3;
  const int wn = wave >> 2;
  const int r  = lane & 15;
  const int hi = lane >> 4;
  const int blockM = blockIdx.y * 128;
  const int blockN = blockIdx.x * 128;

  v8f zero = {};
  v8f acc[2][4];
  for (int mt = 0; mt < 2; ++mt)
    for (int nt = 0; nt < 4; ++nt) acc[mt][nt] = zero;

  gemm_stage(A, Bt, As[0], Bs[0], tid, blockM, blockN, Kdim, 0);
  ASYNC_WAIT();
  __syncthreads();

  const int nk = Kdim >> 5;
  for (int kt = 0; kt < nk; ++kt) {
    const unsigned short* Ac = As[kt & 1];
    const unsigned short* Bc = Bs[kt & 1];
    if (kt + 1 < nk)
      gemm_stage(A, Bt, As[(kt + 1) & 1], Bs[(kt + 1) & 1],
                 tid, blockM, blockN, Kdim, (kt + 1) * 32);

    FragBF afr[2], bfr[4];
    for (int mt = 0; mt < 2; ++mt)
      for (int j = 0; j < 8; ++j) {
        const int kp = ((j < 4) ? (2 * j) : (16 + 2 * (j - 4))) + hi * 8;
        afr[mt].u[j] = *(const unsigned int*)&Ac[(wm * 32 + mt * 16 + r) * GLDA + kp];
      }
    for (int nt = 0; nt < 4; ++nt)
      for (int j = 0; j < 8; ++j) {
        const int kk = ((j < 4) ? (2 * j) : (16 + 2 * (j - 4))) + hi * 8;
        bfr[nt].u[j] = *(const unsigned int*)&Bc[(wn * 64 + nt * 16 + r) * GLDB + kk];
      }
    for (int mt = 0; mt < 2; ++mt)
      for (int nt = 0; nt < 4; ++nt)
        acc[mt][nt] = __builtin_amdgcn_wmma_f32_16x16x32_bf16(
            false, afr[mt].v, false, bfr[nt].v, (short)0, acc[mt][nt], false, false);

    ASYNC_WAIT();
    __syncthreads();
  }

  // ---- epilogue ----
  for (int mt = 0; mt < 2; ++mt)
    for (int nt = 0; nt < 4; ++nt)
      for (int j = 0; j < 8; ++j) {
        const int row = blockM + wm * 32 + mt * 16 + (hi ? (8 + j) : j);
        const int col = blockN + wn * 64 + nt * 16 + r;
        const size_t idx = (size_t)row * N + col;
        float val = acc[mt][nt][j];
        if (cbf16) {
          ((unsigned short*)Cv)[idx] = f2bf(val);
        } else {
          if (R) val += R[idx];
          ((float*)Cv)[idx] = val;
        }
      }
}

// =====================================================================
// Flash attention, causal, H=16, HD=64, T=2048, bf16 in/out, WMMA.
// 128 threads (4 waves); one (b,h) x 64-query block per WG.
// =====================================================================
#define LDT 72

__global__ __launch_bounds__(128)
void attn_kernel(const unsigned short* __restrict__ Qp, const unsigned short* __restrict__ Kp,
                 const unsigned short* __restrict__ Vp, unsigned short* __restrict__ Op) {
  const int T = 2048, H = 16, D = 1024;
  __shared__ unsigned short Qs[64 * LDT];
  __shared__ unsigned short Ks[64 * LDT];
  __shared__ unsigned short Vs[64 * LDT];   // transposed: [d][t]
  __shared__ unsigned short Ps[64 * LDT];
  const int qb = blockIdx.x;
  const int bh = blockIdx.y;
  const int bb = bh / H;
  const int hh = bh % H;
  const int tid  = threadIdx.x;
  const int lane = tid & 31;
  const int wave = tid >> 5;
  const int r  = lane & 15;
  const int hi = lane >> 4;
  const size_t base = (size_t)bb * T * D + (size_t)hh * 64;

  // ---- load Q tile [64][64] bf16 -> LDS ----
  {
    const int c  = (tid & 7) * 8;
    const int r0 = tid >> 3;                     // 0..15
    for (int p = 0; p < 4; ++p) {
      const int row = r0 + p * 16;
      *(us8*)&Qs[row * LDT + c] =
          *(const us8*)(Qp + base + (size_t)(qb * 64 + row) * D + c);
    }
  }

  float m_run[8], l_run[8];
  v8f zero = {};
  v8f oacc[4];
  for (int j = 0; j < 8; ++j) { m_run[j] = -3.0e38f; l_run[j] = 0.0f; }
  for (int nt = 0; nt < 4; ++nt) oacc[nt] = zero;

  for (int kb = 0; kb <= qb; ++kb) {
    __syncthreads();
    // ---- stage K [t][d] and V transposed [d][t] ----
    {
      const int c  = (tid & 7) * 8;
      const int r0 = tid >> 3;
      for (int p = 0; p < 4; ++p) {
        const int row = r0 + p * 16;
        *(us8*)&Ks[row * LDT + c] =
            *(const us8*)(Kp + base + (size_t)(kb * 64 + row) * D + c);
        const us8 vv = *(const us8*)(Vp + base + (size_t)(kb * 64 + row) * D + c);
        for (int i = 0; i < 8; ++i) Vs[(c + i) * LDT + row] = vv[i];
      }
    }
    __syncthreads();

    // ---- S = Q K^T ----
    v8f sacc[4];
    for (int nt = 0; nt < 4; ++nt) sacc[nt] = zero;
    for (int ks = 0; ks < 2; ++ks) {
      FragBF a;
      for (int j = 0; j < 8; ++j) {
        const int kp = ((j < 4) ? (2 * j) : (16 + 2 * (j - 4))) + hi * 8 + ks * 32;
        a.u[j] = *(const unsigned int*)&Qs[(wave * 16 + r) * LDT + kp];
      }
      for (int nt = 0; nt < 4; ++nt) {
        FragBF bf;
        for (int j = 0; j < 8; ++j) {
          const int kk = ((j < 4) ? (2 * j) : (16 + 2 * (j - 4))) + hi * 8 + ks * 32;
          bf.u[j] = *(const unsigned int*)&Ks[(nt * 16 + r) * LDT + kk];
        }
        sacc[nt] = __builtin_amdgcn_wmma_f32_16x16x32_bf16(
            false, a.v, false, bf.v, (short)0, sacc[nt], false, false);
      }
    }

    // ---- online softmax per query row ----
    for (int j = 0; j < 8; ++j) {
      const int rowj = hi ? (8 + j) : j;
      const int tq = qb * 64 + wave * 16 + rowj;
      float sv[4];
      float mloc = -3.0e38f;
      for (int nt = 0; nt < 4; ++nt) {
        float s = sacc[nt][j] * 0.125f;
        const int tk = kb * 64 + nt * 16 + r;
        s = (tk > tq) ? -3.0e38f : s;
        sv[nt] = s;
        mloc = fmaxf(mloc, s);
      }
      for (int m = 1; m < 16; m <<= 1) mloc = fmaxf(mloc, __shfl_xor(mloc, m, 32));
      const float mnew  = fmaxf(m_run[j], mloc);
      const float alpha = __expf(m_run[j] - mnew);
      m_run[j] = mnew;
      float rs = 0.0f;
      for (int nt = 0; nt < 4; ++nt) {
        const float p = __expf(sv[nt] - mnew);
        rs += p;
        Ps[(wave * 16 + rowj) * LDT + nt * 16 + r] = f2bf(p);
      }
      for (int m = 1; m < 16; m <<= 1) rs += __shfl_xor(rs, m, 32);
      l_run[j] = l_run[j] * alpha + rs;
      for (int nt = 0; nt < 4; ++nt) oacc[nt][j] *= alpha;
    }
    __syncthreads();

    // ---- O += P V ----
    for (int ks = 0; ks < 2; ++ks) {
      FragBF a;
      for (int j = 0; j < 8; ++j) {
        const int kp = ((j < 4) ? (2 * j) : (16 + 2 * (j - 4))) + hi * 8 + ks * 32;
        a.u[j] = *(const unsigned int*)&Ps[(wave * 16 + r) * LDT + kp];
      }
      for (int nt = 0; nt < 4; ++nt) {
        FragBF bf;
        for (int j = 0; j < 8; ++j) {
          const int kk = ((j < 4) ? (2 * j) : (16 + 2 * (j - 4))) + hi * 8 + ks * 32;
          bf.u[j] = *(const unsigned int*)&Vs[(nt * 16 + r) * LDT + kk];
        }
        oacc[nt] = __builtin_amdgcn_wmma_f32_16x16x32_bf16(
            false, a.v, false, bf.v, (short)0, oacc[nt], false, false);
      }
    }
  }

  // ---- epilogue: O/l -> bf16 ----
  for (int j = 0; j < 8; ++j) {
    const int rowj = hi ? (8 + j) : j;
    const int tq = qb * 64 + wave * 16 + rowj;
    const float invl = 1.0f / l_run[j];
    for (int nt = 0; nt < 4; ++nt)
      Op[base + (size_t)tq * D + nt * 16 + r] = f2bf(oacc[nt][j] * invl);
  }
}

// =====================================================================
// SwiGLU gate: u1 <- silu(u1) * u2  (bf16 in/out, 8 elems/thread)
// =====================================================================
__global__ __launch_bounds__(256)
void silu_mul_kernel(unsigned short* __restrict__ u1, const unsigned short* __restrict__ u2) {
  const size_t i = ((size_t)blockIdx.x * 256 + threadIdx.x) * 8;
  us8 a = *(us8*)(u1 + i);
  const us8 c = *(const us8*)(u2 + i);
  us8 o;
  for (int e = 0; e < 8; ++e) {
    const float f = bf2f(a[e]);
    o[e] = f2bf(f / (1.0f + __expf(-f)) * bf2f(c[e]));
  }
  *(us8*)(u1 + i) = o;
}

// =====================================================================
extern "C" void kernel_launch(void* const* d_in, const int* in_sizes, int n_in,
                              void* d_out, int out_size, void* d_ws, size_t ws_size,
                              hipStream_t stream) {
  (void)in_sizes; (void)n_in; (void)out_size; (void)ws_size;
  const float* h    = (const float*)d_in[0];
  // d_in[1] = causal mask (implicit in attn_kernel)
  const float* alg  = (const float*)d_in[2];
  const float* alb  = (const float*)d_in[3];
  const float* mlg  = (const float*)d_in[4];
  const float* mlb  = (const float*)d_in[5];
  const float* wq   = (const float*)d_in[6];
  const float* wk   = (const float*)d_in[7];
  const float* wv   = (const float*)d_in[8];
  const float* wo   = (const float*)d_in[9];
  const float* w1   = (const float*)d_in[10];
  const float* w2   = (const float*)d_in[11];
  const float* wout = (const float*)d_in[12];
  float* out = (float*)d_out;

  const size_t MB = 1024u * 1024u;
  char* wsb = (char*)d_ws;
  float*          h1  = (float*)(wsb);                     // 32 MB fp32
  unsigned short* xB  = (unsigned short*)(wsb + 32 * MB);  // 16 MB bf16 (LN out)
  unsigned short* qB  = (unsigned short*)(wsb + 48 * MB);
  unsigned short* kB  = (unsigned short*)(wsb + 64 * MB);
  unsigned short* vB  = (unsigned short*)(wsb + 80 * MB);
  unsigned short* oB  = (unsigned short*)(wsb + 96 * MB);
  unsigned short* u1B = (unsigned short*)(wsb + 112 * MB); // 64 MB
  unsigned short* u2B = (unsigned short*)(wsb + 176 * MB); // 64 MB
  unsigned short* wT  = (unsigned short*)(wsb + 240 * MB); // 32 MB bf16 weights [N][K]
  unsigned short* wqT   = wT;
  unsigned short* wkT   = wT + 1 * MB;
  unsigned short* wvT   = wT + 2 * MB;
  unsigned short* woT   = wT + 3 * MB;
  unsigned short* w1T   = wT + 4 * MB;
  unsigned short* w2T   = wT + 8 * MB;
  unsigned short* woutT = wT + 12 * MB;

  // ---- one-time weight transpose+convert (fp32 [K][N] -> bf16 [N][K]) ----
  transpose_bf16_kernel<<<dim3(16, 16), 256, 0, stream>>>(wq,   wqT,   1024, 1024);
  transpose_bf16_kernel<<<dim3(16, 16), 256, 0, stream>>>(wk,   wkT,   1024, 1024);
  transpose_bf16_kernel<<<dim3(16, 16), 256, 0, stream>>>(wv,   wvT,   1024, 1024);
  transpose_bf16_kernel<<<dim3(16, 16), 256, 0, stream>>>(wo,   woT,   1024, 1024);
  transpose_bf16_kernel<<<dim3(64, 16), 256, 0, stream>>>(w1,   w1T,   1024, 4096);
  transpose_bf16_kernel<<<dim3(64, 16), 256, 0, stream>>>(w2,   w2T,   1024, 4096);
  transpose_bf16_kernel<<<dim3(16, 64), 256, 0, stream>>>(wout, woutT, 4096, 1024);

  const dim3 gOut(8, 64);   // N=1024
  const dim3 gFfn(32, 64);  // N=4096

  ln_kernel<<<8192, 256, 0, stream>>>(h, alg, alb, xB);
  gemm_kernel<<<gOut, 256, 0, stream>>>(xB, wqT, qB, nullptr, 1024, 1024, 1);
  gemm_kernel<<<gOut, 256, 0, stream>>>(xB, wkT, kB, nullptr, 1024, 1024, 1);
  gemm_kernel<<<gOut, 256, 0, stream>>>(xB, wvT, vB, nullptr, 1024, 1024, 1);
  attn_kernel<<<dim3(32, 64), 128, 0, stream>>>(qB, kB, vB, oB);
  gemm_kernel<<<gOut, 256, 0, stream>>>(oB, woT, h1, h, 1024, 1024, 0);
  ln_kernel<<<8192, 256, 0, stream>>>(h1, mlg, mlb, xB);
  gemm_kernel<<<gFfn, 256, 0, stream>>>(xB, w1T, u1B, nullptr, 4096, 1024, 1);
  gemm_kernel<<<gFfn, 256, 0, stream>>>(xB, w2T, u2B, nullptr, 4096, 1024, 1);
  silu_mul_kernel<<<16384, 256, 0, stream>>>(u1B, u2B);
  gemm_kernel<<<gOut, 256, 0, stream>>>(u1B, woutT, out, h1, 1024, 4096, 0);
}
